// Discriminator_28338194219433
// MI455X (gfx1250) — matrix operand order repaired and checked by reading, not compiled, not run on hardware
//
#include <hip/hip_runtime.h>
#include <hip/hip_bf16.h>

// ---------------------------------------------------------------------------
// Neural-CDE discriminator, MI455X (gfx1250, wave32, WMMA).
// Persistent per-batch-tile kernel: each workgroup owns 32 batch rows and
// runs the full 255-step Heun scan locally. All matmuls via
// v_wmma_f32_16x16x32_f16 (f16 in, f32 accum). Layer-3 (128->2112) is fused
// with the einsum contraction (v_tanh_f32 in registers, ds_add_f32 scatter)
// so the (B,2112) activation tensor is never materialized.
//
// Round-5 change:
//  * scatter_tile batches its 8 dx ds_loads before any ds_add_f32: DS loads
//    and DS atomics share the in-order DS queue, so the previous
//    load->add->load interleave forced a dscnt-0 drain per element; now one
//    progressive drain overlaps the tanh chain.
// ---------------------------------------------------------------------------

typedef __attribute__((ext_vector_type(16))) _Float16 v16h;
typedef __attribute__((ext_vector_type(8)))  _Float16 v8h;
typedef __attribute__((ext_vector_type(8)))  float    v8f;

// explicit global-address-space views (keeps weight loads on the
// global_load path regardless of address-space inference)
typedef __attribute__((address_space(1))) const _Float16 g_half;
typedef __attribute__((address_space(1))) const v16h    g_v16h;

#define CHN     33      // 1 + DATA_SIZE
#define HID     64
#define MLPW    128
#define BATCH   4096
#define TLEN    256

#define MT      32      // batch rows per workgroup
#define RT_N    (MT/16) // row tiles
#define NWAVES  8
#define BLOCK   256
#define LDK     96      // padded K for layer-1 input (65 -> 96)
#define CHP     34      // padded dx row stride

#define FRAG    512     // halfs per 32x16 B fragment (32 lanes x 16 halfs)

// fragment tile counts [K-tiles of 32][N-tiles of 16]
#define FW0_KT 3
#define FW0_NT 8        // func W0: 65x128
#define FW1_KT 4
#define FW1_NT 8        // func W1: 128x128
#define FW2_KT 4
#define FW2_NT 132      // func W2: 128x2112
#define IW0_KT 2
#define IW0_NT 8        // init W0: 33x128
#define IW1_KT 4
#define IW1_NT 8        // init W1: 128x128
#define IW2_KT 4
#define IW2_NT 4        // init W2: 128x64

// workspace offsets (in halfs)
#define OFF_FW0 0
#define OFF_FW1 (OFF_FW0 + FW0_KT*FW0_NT*FRAG)
#define OFF_FW2 (OFF_FW1 + FW1_KT*FW1_NT*FRAG)
#define OFF_IW0 (OFF_FW2 + FW2_KT*FW2_NT*FRAG)
#define OFF_IW1 (OFF_IW0 + IW0_KT*IW0_NT*FRAG)
#define OFF_IW2 (OFF_IW1 + IW1_KT*IW1_NT*FRAG)

// ---------------------------------------------------------------------------
// Fast math: gfx1250 v_tanh_f32 / v_rcp_f32 (probe-confirmed in round-3/4 asm).
// ---------------------------------------------------------------------------
__device__ __forceinline__ float fast_rcp(float x)
{
#if __has_builtin(__builtin_amdgcn_rcpf)
    return __builtin_amdgcn_rcpf(x);
#else
    return 1.0f / x;
#endif
}

__device__ __forceinline__ float fast_tanh(float x)
{
#if __has_builtin(__builtin_amdgcn_tanhf)
    return __builtin_amdgcn_tanhf(x);
#else
    float e = __expf(2.0f * x);
    return 1.0f - 2.0f * fast_rcp(e + 1.0f);
#endif
}

__device__ __forceinline__ float lipswish(float x)
{
    return 0.909f * x * fast_rcp(1.0f + __expf(-x));
}

// ---------------------------------------------------------------------------
// Weight pack: f32 (K x N, row-major) -> f16 B-fragment order.
// ISA 16-bit B layout (32x16 tile): element (k,n) lives at
//   lane = n + 16*((k>>3)&1), vgpr = 4*(k>>4) + ((k&7)>>1), half = k&1
// Stored so a wave loads a fragment with one 32B per-lane read.
// ---------------------------------------------------------------------------
__global__ void pack_weights(const float* __restrict__ W, int K, int N,
                             int KT, int NT, _Float16* __restrict__ out)
{
    int total = KT * NT * FRAG;
    for (int o = blockIdx.x * blockDim.x + threadIdx.x; o < total;
         o += gridDim.x * blockDim.x) {
        int kt   = o / (NT * FRAG);
        int rem  = o % (NT * FRAG);
        int nt   = rem / FRAG;
        int r2   = rem % FRAG;
        int lane = r2 >> 4;
        int slot = r2 & 15;
        int v    = slot >> 1;
        int h    = slot & 1;
        int kl   = ((slot >> 3) << 4) + ((lane >> 4) << 3) + (slot & 7);   // k within 32
        int nl   = lane & 15;
        int kg   = kt * 32 + kl;
        int ng   = nt * 16 + nl;
        float val = (kg < K) ? W[(size_t)kg * N + ng] : 0.0f;
        out[o] = (_Float16)val;
    }
}

__global__ void zero_out_kernel(float* p) {
    if (threadIdx.x == 0 && blockIdx.x == 0) p[0] = 0.0f;
}

// ---------------------------------------------------------------------------
// Fragment loads
// ---------------------------------------------------------------------------
// A fragment (16x32 f16, M x K) from LDS row-major [row][k], stride ld.
// ISA layout: lane l holds row (l&15); its 16 halfs are K in
// [kg, kg+8) U [kg+16, kg+24), kg = 8*(l>>4)  -> two ds_load_b128.
__device__ __forceinline__ v16h load_A(const _Float16* X, int ld,
                                       int rowBase, int kt, int lane)
{
    int row = rowBase + (lane & 15);
    int kg  = (lane >> 4) << 3;
    const _Float16* p = X + row * ld + kt * 32 + kg;
    v8h lo = *(const v8h*)p;
    v8h hi = *(const v8h*)(p + 16);
    v16h a;
#pragma unroll
    for (int i = 0; i < 8; ++i) { a[i] = lo[i]; a[i + 8] = hi[i]; }
    return a;
}

// B fragment from pre-packed global array: one 32B per-lane read on the
// explicit global path (global_load_b128, partial s_wait_loadcnt).
__device__ __forceinline__ v16h load_B(const _Float16* frags, int NT,
                                       int kt, int nt, int lane)
{
    g_half* p = (g_half*)frags + ((size_t)(kt * NT + nt)) * FRAG + lane * 16;
    return *(g_v16h*)p;
}

__device__ __forceinline__ v8f wmma_f16(v16h a, v16h b, v8f c)
{
    return __builtin_amdgcn_wmma_f32_16x16x32_f16(
        false, a, false, b, (short)0, c, false, false);
}

// ---------------------------------------------------------------------------
// Dense layer: H = act(X @ W + b), f16 output to LDS.
// B fragments hoisted per column-tile (each wave owns nt = wave, wave+8, ...).
// C/D layout: vgpr v holds (M = v + 8*(lane>=16), N = lane&15).
// All control flow around WMMA is wave-uniform (EXEC all-ones).
// ---------------------------------------------------------------------------
__device__ void dense_act(const _Float16* __restrict__ X, int ldx,
                          const _Float16* __restrict__ Wf, int KT, int NT,
                          const float* __restrict__ bias,
                          _Float16* __restrict__ H, int ldh, int act)
{
    const int lane = threadIdx.x & 31;
    const int wave = threadIdx.x >> 5;
    for (int nt = wave; nt < NT; nt += NWAVES) {
        v16h b[4];
        for (int kt = 0; kt < KT; ++kt) b[kt] = load_B(Wf, NT, kt, nt, lane);
        int   n  = nt * 16 + (lane & 15);
        float bv = bias[n];
#pragma unroll
        for (int rt = 0; rt < RT_N; ++rt) {
            v8f acc = {};
            for (int kt = 0; kt < KT; ++kt) {
                v16h a = load_A(X, ldx, rt * 16, kt, lane);
                acc = wmma_f16(a, b[kt], acc);
            }
            int rbase = rt * 16 + ((lane >> 4) << 3);
#pragma unroll
            for (int v = 0; v < 8; ++v) {
                float x = acc[v] + bv;
                if (act) x = lipswish(x);
                H[(rbase + v) * ldh + n] = (_Float16)x;
            }
        }
    }
}

// Final init layer: linear, f32 output (y state).
__device__ void dense_f32(const _Float16* __restrict__ X, int ldx,
                          const _Float16* __restrict__ Wf, int KT, int NT,
                          const float* __restrict__ bias,
                          float* __restrict__ Y, int ldy)
{
    const int lane = threadIdx.x & 31;
    const int wave = threadIdx.x >> 5;
    for (int nt = wave; nt < NT; nt += NWAVES) {
        v16h b[4];
        for (int kt = 0; kt < KT; ++kt) b[kt] = load_B(Wf, NT, kt, nt, lane);
        int   n  = nt * 16 + (lane & 15);
        float bv = bias[n];
#pragma unroll
        for (int rt = 0; rt < RT_N; ++rt) {
            v8f acc = {};
            for (int kt = 0; kt < KT; ++kt) {
                v16h a = load_A(X, ldx, rt * 16, kt, lane);
                acc = wmma_f16(a, b[kt], acc);
            }
            int rbase = rt * 16 + ((lane >> 4) << 3);
#pragma unroll
            for (int v = 0; v < 8; ++v)
                Y[(rbase + v) * ldy + n] = acc[v] + bv;
        }
    }
}

// tanh + einsum scatter for one 16x16 layer-3 tile held in a C fragment.
// Phase 1: batch all 8 dx ds_loads; phase 2: tanh/mul; phase 3: ds_add_f32.
// (DS loads and DS atomics share the in-order DS queue -- interleaving them
// forces a dscnt-0 drain per element.)
__device__ __forceinline__ void scatter_tile(v8f acc, int nt, int rt, int lane,
                                             const float* __restrict__ b2L,
                                             const float* __restrict__ dxL,
                                             float* __restrict__ facc)
{
    int   g     = nt * 16 + (lane & 15);      // 0..2111
    float bv    = b2L[g];
    int   h     = g / CHN;
    int   c     = g % CHN;
    int   rbase = rt * 16 + ((lane >> 4) << 3);

    float dxv[8];
#pragma unroll
    for (int v = 0; v < 8; ++v)
        dxv[v] = dxL[(rbase + v) * CHP + c];

    float val[8];
#pragma unroll
    for (int v = 0; v < 8; ++v)
        val[v] = fast_tanh(acc[v] + bv) * dxv[v];

#pragma unroll
    for (int v = 0; v < 8; ++v)
        atomicAdd(&facc[(rbase + v) * HID + h], val[v]);   // ds_add_f32
}

// Layer-3 fused with einsum: O = tanh(H2 @ W2 + b2), contracted with dx.
// A fragments (4 x v16h) cached in registers per row-tile; W2 B-fragments
// stream from L2. Per pair: one clause burst of 16 global_load_b128, then
// 8 WMMAs under partial waits, then both scatters (VALU covers next loads).
__device__ void layer3_scatter(const _Float16* __restrict__ H2,
                               const _Float16* __restrict__ W2f,
                               const float* __restrict__ b2L,
                               const float* __restrict__ dxL,
                               float* __restrict__ facc)
{
    const int lane = threadIdx.x & 31;
    const int wave = threadIdx.x >> 5;
#pragma unroll
    for (int rt = 0; rt < RT_N; ++rt) {
        v16h a0 = load_A(H2, MLPW, rt * 16, 0, lane);
        v16h a1 = load_A(H2, MLPW, rt * 16, 1, lane);
        v16h a2 = load_A(H2, MLPW, rt * 16, 2, lane);
        v16h a3 = load_A(H2, MLPW, rt * 16, 3, lane);

        int nt = wave;
        for (; nt + NWAVES < FW2_NT; nt += 2 * NWAVES) {
            int nt2 = nt + NWAVES;
            // issue all 8 B-fragment loads for this pair up front
            v16h b0[4], b1[4];
#pragma unroll
            for (int kt = 0; kt < FW2_KT; ++kt) {
                b0[kt] = load_B(W2f, FW2_NT, kt, nt,  lane);
                b1[kt] = load_B(W2f, FW2_NT, kt, nt2, lane);
            }
            // prefetch next pair (wave-uniform guards)
            if (nt + 2 * NWAVES < FW2_NT) {
#pragma unroll
                for (int kt = 0; kt < FW2_KT; ++kt)
                    __builtin_prefetch(
                        W2f + ((size_t)(kt * FW2_NT + nt + 2 * NWAVES)) * FRAG
                            + lane * 16, 0, 1);
            }
            if (nt + 3 * NWAVES < FW2_NT) {
#pragma unroll
                for (int kt = 0; kt < FW2_KT; ++kt)
                    __builtin_prefetch(
                        W2f + ((size_t)(kt * FW2_NT + nt + 3 * NWAVES)) * FRAG
                            + lane * 16, 0, 1);
            }
            v8f acc0 = {}, acc1 = {};
            acc0 = wmma_f16(a0, b0[0], acc0);
            acc1 = wmma_f16(a0, b1[0], acc1);
            acc0 = wmma_f16(a1, b0[1], acc0);
            acc1 = wmma_f16(a1, b1[1], acc1);
            acc0 = wmma_f16(a2, b0[2], acc0);
            acc1 = wmma_f16(a2, b1[2], acc1);
            acc0 = wmma_f16(a3, b0[3], acc0);
            acc1 = wmma_f16(a3, b1[3], acc1);
            scatter_tile(acc0, nt,  rt, lane, b2L, dxL, facc);
            scatter_tile(acc1, nt2, rt, lane, b2L, dxL, facc);
        }
        if (nt < FW2_NT) {               // tail tile (wave-uniform)
            v8f acc = {};
            acc = wmma_f16(a0, load_B(W2f, FW2_NT, 0, nt, lane), acc);
            acc = wmma_f16(a1, load_B(W2f, FW2_NT, 1, nt, lane), acc);
            acc = wmma_f16(a2, load_B(W2f, FW2_NT, 2, nt, lane), acc);
            acc = wmma_f16(a3, load_B(W2f, FW2_NT, 3, nt, lane), acc);
            scatter_tile(acc, nt, rt, lane, b2L, dxL, facc);
        }
    }
}

// ---------------------------------------------------------------------------
// Main persistent kernel: one workgroup = 32 batch rows, full 255-step scan.
// ---------------------------------------------------------------------------
__global__ __launch_bounds__(BLOCK) void cde_kernel(
    const float* __restrict__ coeffs,
    const float* __restrict__ ib0, const float* __restrict__ ib1,
    const float* __restrict__ ib2,
    const float* __restrict__ fb0, const float* __restrict__ fb1,
    const float* __restrict__ fb2,
    const float* __restrict__ rw,  const float* __restrict__ rb,
    const _Float16* __restrict__ ws, float* __restrict__ out)
{
    __shared__ float sY[MT * HID];
    __shared__ float sZ[MT * HID];
    __shared__ float sF[MT * HID];
    __shared__ float sFacc[MT * HID];
    __shared__ __attribute__((aligned(32))) _Float16 sX[MT * LDK];
    __shared__ __attribute__((aligned(32))) _Float16 sH1[MT * MLPW];
    __shared__ __attribute__((aligned(32))) _Float16 sH2[MT * MLPW];
    __shared__ float sDx[MT * CHP];
    __shared__ float sB0[MLPW];          // func biases cached in LDS
    __shared__ float sB1[MLPW];
    __shared__ float sB2[HID * CHN];
    __shared__ float sRed;

    const int tid = threadIdx.x;
    const int r0  = blockIdx.x * MT;

    const _Float16* fW0 = ws + OFF_FW0;
    const _Float16* fW1 = ws + OFF_FW1;
    const _Float16* fW2 = ws + OFF_FW2;
    const _Float16* iW0 = ws + OFF_IW0;
    const _Float16* iW1 = ws + OFF_IW1;
    const _Float16* iW2 = ws + OFF_IW2;

    // ---- cache func biases + build init input ---------------------------
    for (int o = tid; o < MLPW; o += BLOCK) { sB0[o] = fb0[o]; sB1[o] = fb1[o]; }
    for (int o = tid; o < HID * CHN; o += BLOCK) sB2[o] = fb2[o];
    for (int o = tid; o < MT * LDK; o += BLOCK) {
        int row = o / LDK, k = o % LDK;
        float v = (k < CHN)
                    ? coeffs[((size_t)(r0 + row) * TLEN + 0) * CHN + k] : 0.0f;
        sX[o] = (_Float16)v;
    }
    __syncthreads();
    dense_act(sX, LDK, iW0, IW0_KT, IW0_NT, ib0, sH1, MLPW, 1);
    __syncthreads();
    dense_act(sH1, MLPW, iW1, IW1_KT, IW1_NT, ib1, sH2, MLPW, 1);
    __syncthreads();
    dense_f32(sH2, MLPW, iW2, IW2_KT, IW2_NT, ib2, sY, HID);
    __syncthreads();

    // z0 = y0; dx0 = coeffs[:,1]-coeffs[:,0]
    for (int o = tid; o < MT * HID; o += BLOCK) sZ[o] = sY[o];
    for (int o = tid; o < MT * CHN; o += BLOCK) {
        int row = o / CHN, c = o % CHN;
        size_t base = ((size_t)(r0 + row) * TLEN + 0) * CHN + c;
        sDx[row * CHP + c] = coeffs[base + CHN] - coeffs[base];
    }
    __syncthreads();

    // ---- f0 = vf(0, y0, dx0) --------------------------------------------
    {
        for (int o = tid; o < MT * LDK; o += BLOCK) {
            int row = o / LDK, k = o % LDK;
            float v = (k == 0) ? 0.0f
                    : (k <= HID ? sY[row * HID + (k - 1)] : 0.0f);
            sX[o] = (_Float16)v;
        }
        for (int o = tid; o < MT * HID; o += BLOCK) sFacc[o] = 0.0f;
        __syncthreads();
        dense_act(sX, LDK, fW0, FW0_KT, FW0_NT, sB0, sH1, MLPW, 1);
        __syncthreads();
        dense_act(sH1, MLPW, fW1, FW1_KT, FW1_NT, sB1, sH2, MLPW, 1);
        __syncthreads();
        layer3_scatter(sH2, fW2, sB2, sDx, sFacc);
        __syncthreads();
        for (int o = tid; o < MT * HID; o += BLOCK) sF[o] = sFacc[o];
        __syncthreads();
    }

    // ---- 255-step Heun scan ---------------------------------------------
#pragma unroll 1
    for (int i = 0; i < TLEN - 1; ++i) {
        // z1 = 2y - z + f  (in place)
        for (int o = tid; o < MT * HID; o += BLOCK)
            sZ[o] = 2.0f * sY[o] - sZ[o] + sF[o];
        // dx1 = dX[:, min(i+1, T-2)]
        int j = (i + 1 < TLEN - 1) ? (i + 1) : (TLEN - 2);
        for (int o = tid; o < MT * CHN; o += BLOCK) {
            int row = o / CHN, c = o % CHN;
            size_t base = ((size_t)(r0 + row) * TLEN + j) * CHN + c;
            sDx[row * CHP + c] = coeffs[base + CHN] - coeffs[base];
        }
        __syncthreads();

        // f1 = vf(i+1, z1, dx1)
        float t1 = (float)(i + 1);
        for (int o = tid; o < MT * LDK; o += BLOCK) {
            int row = o / LDK, k = o % LDK;
            float v = (k == 0) ? t1
                    : (k <= HID ? sZ[row * HID + (k - 1)] : 0.0f);
            sX[o] = (_Float16)v;
        }
        for (int o = tid; o < MT * HID; o += BLOCK) sFacc[o] = 0.0f;
        __syncthreads();
        dense_act(sX, LDK, fW0, FW0_KT, FW0_NT, sB0, sH1, MLPW, 1);
        __syncthreads();
        dense_act(sH1, MLPW, fW1, FW1_KT, FW1_NT, sB1, sH2, MLPW, 1);
        __syncthreads();
        layer3_scatter(sH2, fW2, sB2, sDx, sFacc);
        __syncthreads();

        // y += 0.5*(f + f1); f = f1
        for (int o = tid; o < MT * HID; o += BLOCK) {
            float f1 = sFacc[o];
            sY[o] += 0.5f * (sF[o] + f1);
            sF[o]  = f1;
        }
        __syncthreads();
    }

    // ---- readout + mean reduction ---------------------------------------
    if (tid == 0) sRed = 0.0f;
    __syncthreads();
    float part = 0.0f;
    for (int row = tid; row < MT; row += BLOCK) {   // only tid < 32 work
        float s = rb[0];
        for (int h = 0; h < HID; ++h) s += sY[row * HID + h] * rw[h];
        part += s;
    }
    atomicAdd(&sRed, part);
    __syncthreads();
    if (tid == 0) atomicAdd(out, sRed * (1.0f / (float)BATCH));
}

// ---------------------------------------------------------------------------
// Launch. d_in order (setup_inputs dict, nested params flattened in order):
//  [0] coeffs            [1] iW0  [2] ib0  [3] iW1  [4] ib1  [5] iW2  [6] ib2
//  [7] fW0  [8] fb0  [9] fW1 [10] fb1 [11] fW2 [12] fb2
// [13] readout_w [14] readout_b
// ---------------------------------------------------------------------------
extern "C" void kernel_launch(void* const* d_in, const int* in_sizes, int n_in,
                              void* d_out, int out_size, void* d_ws,
                              size_t ws_size, hipStream_t stream)
{
    (void)in_sizes; (void)n_in; (void)out_size; (void)ws_size;

    const float* coeffs = (const float*)d_in[0];
    const float* iW0 = (const float*)d_in[1];
    const float* ib0 = (const float*)d_in[2];
    const float* iW1 = (const float*)d_in[3];
    const float* ib1 = (const float*)d_in[4];
    const float* iW2 = (const float*)d_in[5];
    const float* ib2 = (const float*)d_in[6];
    const float* fW0 = (const float*)d_in[7];
    const float* fb0 = (const float*)d_in[8];
    const float* fW1 = (const float*)d_in[9];
    const float* fb1 = (const float*)d_in[10];
    const float* fW2 = (const float*)d_in[11];
    const float* fb2 = (const float*)d_in[12];
    const float* rw  = (const float*)d_in[13];
    const float* rb  = (const float*)d_in[14];

    _Float16* ws  = (_Float16*)d_ws;
    float*    out = (float*)d_out;

    zero_out_kernel<<<1, 64, 0, stream>>>(out);

    auto pack = [&](const float* W, int K, int N, int KT, int NT, int off) {
        int total  = KT * NT * FRAG;
        int blocks = (total + 255) / 256;
        pack_weights<<<blocks, 256, 0, stream>>>(W, K, N, KT, NT, ws + off);
    };
    pack(fW0,  65, MLPW,       FW0_KT, FW0_NT, OFF_FW0);
    pack(fW1, 128, MLPW,       FW1_KT, FW1_NT, OFF_FW1);
    pack(fW2, 128, HID * CHN,  FW2_KT, FW2_NT, OFF_FW2);
    pack(iW0,  33, MLPW,       IW0_KT, IW0_NT, OFF_IW0);
    pack(iW1, 128, MLPW,       IW1_KT, IW1_NT, OFF_IW1);
    pack(iW2, 128, HID,        IW2_KT, IW2_NT, OFF_IW2);

    cde_kernel<<<BATCH / MT, BLOCK, 0, stream>>>(
        coeffs, ib0, ib1, ib2, fb0, fb1, fb2, rw, rb, ws, out);
}
